// MultiheadCrossAttention_3590592660074
// MI455X (gfx1250) — matrix operand order repaired and checked by reading, not compiled, not run on hardware
//
#include <hip/hip_runtime.h>

#define B_   4
#define SQ_  1024
#define SKV_ 1024
#define D_   1024
#define H_   16
#define HD_  64

typedef __attribute__((ext_vector_type(16))) __bf16 v16bf;
typedef __attribute__((ext_vector_type(8)))  float  v8f;

struct __align__(16) F128 { float x, y, z, w; };
struct __align__(16) U128 { unsigned int x, y, z, w; };

union BF16Frag {
  v16bf v;
  U128  q[2];
  unsigned short u[16];
};

__device__ __forceinline__ unsigned short f2bf(float f) {
  __bf16 h = (__bf16)f;                       // hardware cvt on gfx1250
  return __builtin_bit_cast(unsigned short, h);
}

__device__ __forceinline__ void cvt8(unsigned short* d, const F128& a, const F128& b) {
  d[0] = f2bf(a.x); d[1] = f2bf(a.y); d[2] = f2bf(a.z); d[3] = f2bf(a.w);
  d[4] = f2bf(b.x); d[5] = f2bf(b.y); d[6] = f2bf(b.z); d[7] = f2bf(b.w);
}

// A-fragment (16x32, M x K): lane r holds row r; lanes 0-15 K {0..7,16..23},
// lanes 16-31 K {8..15,24..31}.  klo = (lane>=16)?8:0.
__device__ __forceinline__ void load_fragA_bf16(BF16Frag& f, const unsigned short* p, int klo) {
  f.q[0] = *(const U128*)(p + klo);
  f.q[1] = *(const U128*)(p + klo + 16);
}

// B-fragment (32x16, K x N): lane r holds column r; lanes 0-15 K {0..7,16..23},
// lanes 16-31 K {8..15,24..31}.  kb = (lane>=16)?16:0.
__device__ __forceinline__ void load_fragB_bf16(BF16Frag& f, const unsigned short* p, int kb) {
  f.q[0] = *(const U128*)(p + (kb >> 1));
  f.q[1] = *(const U128*)(p + (kb >> 1) + 16);
}

__device__ __forceinline__ v8f wmma_bf16(const BF16Frag& a, const BF16Frag& b, v8f c) {
  return __builtin_amdgcn_wmma_f32_16x16x32_bf16(false, a.v, false, b.v, (short)0, c,
                                                 false, false);
}

__device__ __forceinline__ void async_b128(unsigned lds, const unsigned short* g) {
  asm volatile("global_load_async_to_lds_b128 %0, %1, off"
               :: "v"(lds), "v"(g) : "memory");
}
__device__ __forceinline__ void wait_async0() {
  asm volatile("s_wait_asynccnt 0x0" ::: "memory");
}

// ---------------------------------------------------------------------------
// Stage 0: one-shot f32 -> bf16 conversion (8 elems / thread).
// ---------------------------------------------------------------------------
__global__ __launch_bounds__(256) void cvt_bf16_kernel(const float* __restrict__ s,
                                                       unsigned short* __restrict__ d,
                                                       int n) {
  const int i = (blockIdx.x * 256 + threadIdx.x) * 8;
  if (i >= n) return;
  F128 a = *(const F128*)(s + i);
  F128 b = *(const F128*)(s + i + 4);
  union { unsigned short t[8]; U128 u; } x;
  cvt8(x.t, a, b);
  *(U128*)(d + i) = x.u;
}

// ---------------------------------------------------------------------------
// Stage 1: fused projection (X @ W^T + b) + per-head LayerNorm, emit bf16.
// W tile (64 cols x 32 k = 4KB) double-buffered in LDS via async-to-LDS so the
// fill overlaps with WMMA on the other buffer.  All four B-tiles are held in
// distinct fragment registers so ds_loads clause together and drain with
// graduated dscnt waits under the WMMAs.
//   proj 0 -> Qn [B][H][SQ][HD], proj 1 -> Kn, proj 2 -> Vt [B][H][HD][SKV].
// ---------------------------------------------------------------------------
__global__ __launch_bounds__(128) void proj_ln_kernel(
    const unsigned short* __restrict__ xq, const unsigned short* __restrict__ xkv,
    const unsigned short* __restrict__ Wqb, const unsigned short* __restrict__ Wkb,
    const unsigned short* __restrict__ Wvb,
    const float* __restrict__ bq, const float* __restrict__ bk,
    const float* __restrict__ bv,
    const float* __restrict__ gamma, const float* __restrict__ beta,
    unsigned short* __restrict__ qn, unsigned short* __restrict__ kn,
    unsigned short* __restrict__ vt)
{
  __shared__ unsigned short wtile[2][64 * 32];   // double-buffered, 2 x 4 KB

  const int tid  = threadIdx.x;
  const int lane = tid & 31;
  const int wave = tid >> 5;
  const int proj = blockIdx.z;
  const int h    = blockIdx.y;
  const int rt   = blockIdx.x;

  const unsigned short* X = (proj == 0) ? xq : xkv;
  const unsigned short* W = (proj == 0) ? Wqb : ((proj == 1) ? Wkb : Wvb);
  const float* bias       = (proj == 0) ? bq : ((proj == 1) ? bk : bv);

  const int r   = lane & 15;
  const int hi  = lane >> 4;
  const int klo = hi << 3;   // A-fragment K split
  const int kb  = hi << 4;   // B-fragment K split

  const long grow = (long)rt * 64 + wave * 16 + r;
  const unsigned short* Xrow = X + grow * D_;

  // Async staging: 256 16B chunks/step; thread t moves chunks 2t, 2t+1.
  // chunk c -> W row (c>>2), k-segment (c&3)*8.
  const int ca = tid * 2, cb = ca + 1;
  const unsigned short* wga = W + (long)(h * 64 + (ca >> 2)) * D_ + (ca & 3) * 8;
  const unsigned short* wgb = W + (long)(h * 64 + (cb >> 2)) * D_ + (cb & 3) * 8;
  const unsigned ldsa = (unsigned)(uintptr_t)&wtile[0][(ca >> 2) * 32 + (ca & 3) * 8];
  const unsigned ldsb = (unsigned)(uintptr_t)&wtile[0][(cb >> 2) * 32 + (cb & 3) * 8];

  v8f c0 = {}, c1 = {}, c2 = {}, c3 = {};

  // prologue: fill buffer 0
  async_b128(ldsa, wga);
  async_b128(ldsb, wgb);

  int sel = 0;
  for (int k0 = 0; k0 < D_; k0 += 32) {
    wait_async0();                 // current buffer's fill complete (own wave)
    __syncthreads();               // visible to all waves; prev readers done
    if (k0 + 32 < D_) {            // overlap: fill other buffer during WMMA
      const unsigned off = sel ? 0u : 4096u;
      async_b128(ldsa + off, wga + k0 + 32);
      async_b128(ldsb + off, wgb + k0 + 32);
    }
    const unsigned short* wb = &wtile[sel][0];
    BF16Frag a, b0, b1, b2, b3;
    load_fragA_bf16(a, Xrow + k0, klo);
    load_fragB_bf16(b0, wb + ( 0 + r) * 32, kb);
    load_fragB_bf16(b1, wb + (16 + r) * 32, kb);
    load_fragB_bf16(b2, wb + (32 + r) * 32, kb);
    load_fragB_bf16(b3, wb + (48 + r) * 32, kb);
    c0 = wmma_bf16(a, b0, c0);
    c1 = wmma_bf16(a, b1, c1);
    c2 = wmma_bf16(a, b2, c2);
    c3 = wmma_bf16(a, b3, c3);
    sel ^= 1;
  }

  const float bb0 = bias[h * 64 +  0 + r];
  const float bb1 = bias[h * 64 + 16 + r];
  const float bb2 = bias[h * 64 + 32 + r];
  const float bb3 = bias[h * 64 + 48 + r];
  const float g0 = gamma[ 0 + r], g1 = gamma[16 + r];
  const float g2 = gamma[32 + r], g3 = gamma[48 + r];
  const float e0 = beta[ 0 + r],  e1 = beta[16 + r];
  const float e2 = beta[32 + r],  e3 = beta[48 + r];

#pragma unroll
  for (int j = 0; j < 8; ++j) {
    float x0 = c0[j] + bb0, x1 = c1[j] + bb1, x2 = c2[j] + bb2, x3 = c3[j] + bb3;
    float s1 = x0 + x1 + x2 + x3;
    float s2 = x0 * x0 + x1 * x1 + x2 * x2 + x3 * x3;
#pragma unroll
    for (int mk = 8; mk >= 1; mk >>= 1) {
      s1 += __shfl_xor(s1, mk, 32);
      s2 += __shfl_xor(s2, mk, 32);
    }
    const float mu   = s1 * (1.0f / 64.0f);
    const float var  = s2 * (1.0f / 64.0f) - mu * mu;
    const float rstd = rsqrtf(fmaxf(var, 0.0f) + 1e-5f);
    const float y0 = (x0 - mu) * rstd * g0 + e0;
    const float y1 = (x1 - mu) * rstd * g1 + e1;
    const float y2 = (x2 - mu) * rstd * g2 + e2;
    const float y3 = (x3 - mu) * rstd * g3 + e3;

    const int rowj = rt * 64 + wave * 16 + j + (hi << 3);
    const int bidx = rowj >> 10;
    const int s    = rowj & (SQ_ - 1);
    const long bh  = (long)bidx * H_ + h;

    if (proj == 2) {
      vt[(bh * HD_ +  0 + r) * (long)SKV_ + s] = f2bf(y0);
      vt[(bh * HD_ + 16 + r) * (long)SKV_ + s] = f2bf(y1);
      vt[(bh * HD_ + 32 + r) * (long)SKV_ + s] = f2bf(y2);
      vt[(bh * HD_ + 48 + r) * (long)SKV_ + s] = f2bf(y3);
    } else {
      unsigned short* dst = ((proj == 0) ? qn : kn) + (bh * SQ_ + s) * (long)HD_;
      dst[ 0 + r] = f2bf(y0);
      dst[16 + r] = f2bf(y1);
      dst[32 + r] = f2bf(y2);
      dst[48 + r] = f2bf(y3);
    }
  }
}

// ---------------------------------------------------------------------------
// Stage 2: flash attention per (b,h).  Block: 256 threads (8 waves), each wave
// owns 16 q-rows x HD=64 output; streams SKV in chunks of 32.
// Shared K chunk (32x64) and Vt chunk (64x32) double-buffered in LDS via
// async-to-LDS.  K/V fragments in distinct registers for clause'd ds_loads.
// Masked scores are ZERO (not -inf): reference multiplies by mask pre-softmax.
// ---------------------------------------------------------------------------
__global__ __launch_bounds__(256) void attn_kernel(
    const unsigned short* __restrict__ qn, const unsigned short* __restrict__ kn,
    const unsigned short* __restrict__ vt,
    const int* __restrict__ seq_len, const int* __restrict__ kv_seq_len,
    float* __restrict__ out)
{
  __shared__ unsigned short plds[8][16 * 32];    // per-wave C->A relayout, 8 KB
  __shared__ unsigned short kbuf[2][32 * 64];    // K chunk [kv][d], 2 x 4 KB
  __shared__ unsigned short vbuf[2][64 * 32];    // Vt chunk [d][kv], 2 x 4 KB

  const int tid  = threadIdx.x;
  const int lane = tid & 31;
  const int wave = tid >> 5;
  const int bh   = blockIdx.y;
  const int b    = bh >> 4;
  const int h    = bh & 15;
  const int qbase = blockIdx.x * 128 + wave * 16;
  const int r   = lane & 15;
  const int hi  = lane >> 4;
  const int klo = hi << 3;
  const int kb  = hi << 4;

  const int qlen  = seq_len[b];
  const int kvlen = kv_seq_len[b];

  const unsigned short* kbase = kn + (long)bh * SKV_ * HD_;
  const unsigned short* vbase = vt + (long)bh * HD_ * SKV_;

  // Async staging: thread t moves one 16B K chunk and one 16B V chunk per step.
  // K: 32 rows x 64 shorts -> 8 chunks/row: row = t>>3, seg = t&7.
  // V: 64 rows x 32 shorts -> 4 chunks/row: row = t>>2, seg = t&3.
  const unsigned short* kg0 = kbase + (long)(tid >> 3) * HD_  + (tid & 7) * 8;
  const unsigned short* vg0 = vbase + (long)(tid >> 2) * SKV_ + (tid & 3) * 8;
  const unsigned klds = (unsigned)(uintptr_t)&kbuf[0][(tid >> 3) * 64 + (tid & 7) * 8];
  const unsigned vlds = (unsigned)(uintptr_t)&vbuf[0][(tid >> 2) * 32 + (tid & 3) * 8];

  const unsigned short* qrow = qn + ((long)bh * SQ_ + qbase + r) * HD_;
  BF16Frag aq0, aq1;
  load_fragA_bf16(aq0, qrow, klo);
  load_fragA_bf16(aq1, qrow + 32, klo);

  v8f o0 = {}, o1 = {}, o2 = {}, o3 = {};
  float m[8], l[8];
  bool rv[8];
#pragma unroll
  for (int j = 0; j < 8; ++j) {
    m[j] = -1e30f; l[j] = 0.0f;
    rv[j] = (qbase + j + (hi << 3)) < qlen;
  }

  unsigned short* pw = plds[wave];

  // prologue: fill buffer 0 with chunk kc=0
  async_b128(klds, kg0);
  async_b128(vlds, vg0);

  int sel = 0;
  for (int kc = 0; kc < SKV_; kc += 32) {
    wait_async0();
    __syncthreads();                        // K/V chunk visible to all waves
    if (kc + 32 < SKV_) {                   // overlap next fill with compute
      const unsigned off = sel ? 0u : 4096u;
      async_b128(klds + off, kg0 + (long)(kc + 32) * HD_);
      async_b128(vlds + off, vg0 + kc + 32);
    }
    const unsigned short* kbl = &kbuf[sel][0];
    const unsigned short* vbl = &vbuf[sel][0];

    // ---- scores: S[16 x 32] = Q(16x64) @ K(kc..kc+31, 64)^T
    BF16Frag bk0, bk1, bk2, bk3;
    load_fragB_bf16(bk0, kbl + ( 0 + r) * 64,      kb);
    load_fragB_bf16(bk1, kbl + ( 0 + r) * 64 + 32, kb);
    load_fragB_bf16(bk2, kbl + (16 + r) * 64,      kb);
    load_fragB_bf16(bk3, kbl + (16 + r) * 64 + 32, kb);
    v8f s0 = {}, s1 = {};
    s0 = wmma_bf16(aq0, bk0, s0);
    s0 = wmma_bf16(aq1, bk1, s0);
    s1 = wmma_bf16(aq0, bk2, s1);
    s1 = wmma_bf16(aq1, bk3, s1);

    const bool cv0 = (kc + r) < kvlen;
    const bool cv1 = (kc + 16 + r) < kvlen;

    // ---- online softmax (row stats per 16-lane half via shfl_xor)
    float al[8];
#pragma unroll
    for (int j = 0; j < 8; ++j) {
      float a = (cv0 && rv[j]) ? s0[j] : 0.0f;
      float c = (cv1 && rv[j]) ? s1[j] : 0.0f;
      float t = fmaxf(a, c);
#pragma unroll
      for (int mk = 8; mk >= 1; mk >>= 1) t = fmaxf(t, __shfl_xor(t, mk, 32));
      const float mn    = fmaxf(m[j], t);
      const float alpha = __expf(m[j] - mn);
      const float p0 = __expf(a - mn);
      const float p1 = __expf(c - mn);
      float rs = p0 + p1;
#pragma unroll
      for (int mk = 8; mk >= 1; mk >>= 1) rs += __shfl_xor(rs, mk, 32);
      m[j]  = mn;
      l[j]  = l[j] * alpha + rs;
      al[j] = alpha;
      const int row = j + (hi << 3);
      pw[row * 32 + r]      = f2bf(p0);   // C-layout -> LDS (row-major 16x32)
      pw[row * 32 + 16 + r] = f2bf(p1);
    }
#pragma unroll
    for (int j = 0; j < 8; ++j) {
      o0[j] *= al[j]; o1[j] *= al[j]; o2[j] *= al[j]; o3[j] *= al[j];
    }
    __syncthreads();                        // plds ready (and K reads done)

    // ---- O(16x64) += P(16x32) @ Vt(32 x 64)
    BF16Frag pa, vb0, vb1, vb2, vb3;
    load_fragA_bf16(pa, pw + r * 32, klo);
    load_fragB_bf16(vb0, vbl + ( 0 + r) * 32, kb);
    load_fragB_bf16(vb1, vbl + (16 + r) * 32, kb);
    load_fragB_bf16(vb2, vbl + (32 + r) * 32, kb);
    load_fragB_bf16(vb3, vbl + (48 + r) * 32, kb);
    o0 = wmma_bf16(pa, vb0, o0);
    o1 = wmma_bf16(pa, vb1, o1);
    o2 = wmma_bf16(pa, vb2, o2);
    o3 = wmma_bf16(pa, vb3, o3);
    sel ^= 1;
  }

  // ---- epilogue: divide by softmax denominator, write f32 output
#pragma unroll
  for (int j = 0; j < 8; ++j) {
    const float inv = 1.0f / l[j];   // l >= 1 always (all SKV columns counted)
    const int srow = qbase + j + (hi << 3);
    float* dst = out + ((long)b * SQ_ + srow) * D_ + h * HD_;
    dst[ 0 + r] = o0[j] * inv;
    dst[16 + r] = o1[j] * inv;
    dst[32 + r] = o2[j] * inv;
    dst[48 + r] = o3[j] * inv;
  }
}

extern "C" void kernel_launch(void* const* d_in, const int* in_sizes, int n_in,
                              void* d_out, int out_size, void* d_ws, size_t ws_size,
                              hipStream_t stream) {
  (void)in_sizes; (void)n_in; (void)out_size; (void)ws_size;

  const float* q     = (const float*)d_in[0];
  const float* kv    = (const float*)d_in[1];
  const int*   seq   = (const int*)d_in[2];
  const int*   kvseq = (const int*)d_in[3];
  const float* Wq    = (const float*)d_in[4];
  const float* bq    = (const float*)d_in[5];
  const float* Wk    = (const float*)d_in[6];
  const float* bk    = (const float*)d_in[7];
  const float* Wv    = (const float*)d_in[8];
  const float* bvp   = (const float*)d_in[9];
  const float* gamma = (const float*)d_in[10];
  const float* beta  = (const float*)d_in[11];
  float* out = (float*)d_out;

  // Workspace (bf16 elements):
  //   Qn | Kn | Vt : 4M each   (stage-1 outputs)
  //   qb | kvb     : 4M each   (bf16 copies of q, kv)
  //   wqb|wkb|wvb  : 1M each   (bf16 copies of weights)     total 46 MB
  const size_t NTOK = (size_t)B_ * H_ * SQ_ * HD_;   // 4M
  const size_t NW   = (size_t)D_ * D_;               // 1M
  unsigned short* qn  = (unsigned short*)d_ws;
  unsigned short* kn  = qn  + NTOK;
  unsigned short* vt  = kn  + NTOK;
  unsigned short* qb  = vt  + NTOK;
  unsigned short* kvb = qb  + NTOK;
  unsigned short* wqb = kvb + NTOK;
  unsigned short* wkb = wqb + NW;
  unsigned short* wvb = wkb + NW;

  const int ntok = (int)NTOK, nw = (int)NW;
  cvt_bf16_kernel<<<dim3(ntok / 8 / 256), dim3(256), 0, stream>>>(q,  qb,  ntok);
  cvt_bf16_kernel<<<dim3(ntok / 8 / 256), dim3(256), 0, stream>>>(kv, kvb, ntok);
  cvt_bf16_kernel<<<dim3(nw / 8 / 256),   dim3(256), 0, stream>>>(Wq, wqb, nw);
  cvt_bf16_kernel<<<dim3(nw / 8 / 256),   dim3(256), 0, stream>>>(Wk, wkb, nw);
  cvt_bf16_kernel<<<dim3(nw / 8 / 256),   dim3(256), 0, stream>>>(Wv, wvb, nw);

  dim3 g1((B_ * SQ_) / 64, H_, 3), b1(128);
  proj_ln_kernel<<<g1, b1, 0, stream>>>(qb, kvb, wqb, wkb, wvb, bq, bk, bvp,
                                        gamma, beta, qn, kn, vt);

  dim3 g2(SQ_ / 128, B_ * H_), b2(256);
  attn_kernel<<<g2, b2, 0, stream>>>(qn, kn, vt, seq, kvseq, out);
}
